// DyIntraModalityUpdate_36197984371018
// MI455X (gfx1250) — matrix operand order repaired and checked by reading, not compiled, not run on hardware
//
#include <hip/hip_runtime.h>
#include <cstdint>
#include <cstddef>

// Problem constants (match reference)
#define B_   16
#define N_   768
#define D_   512
#define OUT_ 512
#define H_   8
#define DH_  64
#define MN_  (B_ * N_)    // 12288 rows per modality
#define TR_  (3 * OUT_)   // 1536

#define AS1 __attribute__((address_space(1)))
#define AS3 __attribute__((address_space(3)))

typedef __attribute__((ext_vector_type(16))) __bf16       v16bf;
typedef __attribute__((ext_vector_type(8)))  float        v8f;
typedef __attribute__((ext_vector_type(4)))  unsigned int u32x4;
typedef __attribute__((ext_vector_type(8)))  unsigned int u32x8;
typedef __attribute__((ext_vector_type(4)))  int          i32x4;

union BF16Frag { v16bf v; u32x4 u[2]; };

// ---- CDNA5 async/TDM feature detection (compile-safe fallbacks) -----------
#if defined(__has_builtin)
#if __has_builtin(__builtin_amdgcn_global_load_async_to_lds_b128)
#define HAVE_ASYNC_LDS 1
#endif
#endif
#ifndef HAVE_ASYNC_LDS
#define HAVE_ASYNC_LDS 0
#endif

// 16B global -> LDS copy (async where available; ASYNCcnt-tracked)
static __device__ __forceinline__ void async_copy16(const void* g, void* l) {
#if HAVE_ASYNC_LDS
  __builtin_amdgcn_global_load_async_to_lds_b128(
      (AS1 i32x4*)(AS1 void*)(void*)g, (AS3 i32x4*)(AS3 void*)l, 0, 0);
#else
  *(u32x4*)l = *(const u32x4*)g;
#endif
}

static __device__ __forceinline__ void wait_async() {
#if HAVE_ASYNC_LDS
#if __has_builtin(__builtin_amdgcn_s_wait_asynccnt)
  __builtin_amdgcn_s_wait_asynccnt(0);
#else
  asm volatile("s_wait_asynccnt 0x0" ::: "memory");
#endif
#endif
}

static __device__ __forceinline__ void wait_tensor() {
#if __has_builtin(__builtin_amdgcn_s_wait_tensorcnt)
  __builtin_amdgcn_s_wait_tensorcnt((short)0);
#else
  asm volatile("s_wait_tensorcnt 0x0" ::: "memory");
#endif
}

// Tensor Data Mover: load a 2D tile (16-bit elements) into LDS.
//   tile_d0 elements per row, tile_d1 rows, row stride stride_d0 (elements).
// D# per CDNA5 ISA; issued wave-uniform.
static __device__ __forceinline__ void tdm_load_tile_2d(
    unsigned lds_off, unsigned long long gaddr, unsigned tile_d0,
    unsigned tile_d1, unsigned tensor_d0, unsigned tensor_d1,
    unsigned long long stride_d0) {
  u32x4 g0;
  g0[0] = 0x1u;  // count=1, is_restore=0, gather off
  g0[1] = lds_off;
  g0[2] = (unsigned)(gaddr & 0xFFFFFFFFull);
  g0[3] = (unsigned)((gaddr >> 32) & 0x01FFFFFFull) | (2u << 30);  // type=2
  u32x8 g1;
  g1[0] = 1u << 16;  // workgroup_mask=0, data_size=1 (2 bytes)
  g1[1] = (tensor_d0 & 0xFFFFu) << 16;                        // dim0 lo
  g1[2] = ((tensor_d0 >> 16) & 0xFFFFu) | ((tensor_d1 & 0xFFFFu) << 16);
  g1[3] = ((tensor_d1 >> 16) & 0xFFFFu) | (tile_d0 << 16);    // tile_dim0
  g1[4] = tile_d1 & 0xFFFFu;                                  // tile_dim1
  g1[5] = (unsigned)(stride_d0 & 0xFFFFFFFFull);              // dim0 stride
  g1[6] = (unsigned)((stride_d0 >> 32) & 0xFFFFull);
  g1[7] = 0u;
  asm volatile("tensor_load_to_lds %0, %1" ::"s"(g0), "s"(g1) : "memory");
}

static __device__ __forceinline__ unsigned short f2bf(float f) {
  unsigned u = __builtin_bit_cast(unsigned, f);
  unsigned r = 0x7FFFu + ((u >> 16) & 1u);  // round-to-nearest-even
  return (unsigned short)((u + r) >> 16);
}

static __device__ __forceinline__ v8f vzero8() {
  v8f z = {0.f, 0.f, 0.f, 0.f, 0.f, 0.f, 0.f, 0.f};
  return z;
}

static __device__ __forceinline__ v8f wmma_bf16(BF16Frag a, BF16Frag b, v8f c) {
  return __builtin_amdgcn_wmma_f32_16x16x32_bf16(false, a.v, false, b.v,
                                                 (short)0, c, false, false);
}

// ---------------------------------------------------------------------------
// f32 -> bf16 conversion
// ---------------------------------------------------------------------------
__global__ void cvt_bf16_kernel(const float* __restrict__ in,
                                unsigned short* __restrict__ out, int n) {
  int i = blockIdx.x * 256 + threadIdx.x;
  if (i < n) out[i] = f2bf(in[i]);
}

__global__ void add_cvt_bf16_kernel(const float* __restrict__ a,
                                    const float* __restrict__ b,
                                    unsigned short* __restrict__ out, int n) {
  int i = blockIdx.x * 256 + threadIdx.x;
  if (i < n) out[i] = f2bf(a[i] + b[i]);
}

// ---------------------------------------------------------------------------
// Masked means: means[mod][b][d] = sum_n x*mask / sum_n mask ; grid (B_,2)
// ---------------------------------------------------------------------------
__global__ void means_kernel(const float* __restrict__ v, const float* __restrict__ q,
                             const float* __restrict__ vm, const float* __restrict__ qm,
                             float* __restrict__ means) {
  int b = blockIdx.x, mod = blockIdx.y;
  const float* x  = mod ? q : v;
  const float* mk = (mod ? qm : vm) + (size_t)b * N_;
  __shared__ float red[256];
  float s = 0.f;
  for (int n = threadIdx.x; n < N_; n += 256) s += mk[n];
  red[threadIdx.x] = s;
  __syncthreads();
  for (int off = 128; off > 0; off >>= 1) {
    if (threadIdx.x < off) red[threadIdx.x] += red[threadIdx.x + off];
    __syncthreads();
  }
  float inv = 1.0f / fmaxf(red[0], 1e-20f);
  for (int d = threadIdx.x; d < D_; d += 256) {
    const float* xp = x + (size_t)b * N_ * D_ + d;
    float acc = 0.f;
    for (int n = 0; n < N_; ++n) acc += xp[(size_t)n * D_] * mk[n];
    means[((size_t)mod * B_ + b) * D_ + d] = acc * inv;
  }
}

// ---------------------------------------------------------------------------
// Gate factors: gmul[0]=1+sig(q_mean.w_q4v+b) gates v; gmul[1] gates q.
// ---------------------------------------------------------------------------
__global__ void gates_kernel(const float* __restrict__ means,
                             const float* __restrict__ w_v4q, const float* __restrict__ b_v4q,
                             const float* __restrict__ w_q4v, const float* __restrict__ b_q4v,
                             float* __restrict__ gmul) {
  int b = blockIdx.x, mod = blockIdx.y, c = threadIdx.x;
  const float* mean = means + ((size_t)(mod ? 0 : 1) * B_ + b) * D_;
  const float* w    = (mod ? w_v4q : w_q4v) + (size_t)c * D_;
  float acc = (mod ? b_v4q : b_q4v)[c];
  for (int k = 0; k < D_; ++k) acc += mean[k] * w[k];
  gmul[((size_t)mod * B_ + b) * OUT_ + c] = 1.0f + 1.0f / (1.0f + expf(-acc));
}

// ---------------------------------------------------------------------------
// bf16 WMMA GEMM: out[row,col] = A[row,:].W[col,:] + bias[col]
//   epilogue: *=mask[row], *=gate[(row/batchRows)*512 + col%512]; bf16|f32 out
// 256 threads, tile 128x128, K step 32. W tile staged by TDM (wave 0),
// A tile by per-lane async LDS copies.
// ---------------------------------------------------------------------------
__global__ __launch_bounds__(256) void gemm_wmma_kernel(
    const unsigned short* __restrict__ A, const unsigned short* __restrict__ W,
    const float* __restrict__ bias, const float* __restrict__ gate,
    const float* __restrict__ mask, float* __restrict__ outF,
    unsigned short* __restrict__ outH, int M, int Ncol, int K, int batchRows) {
  __shared__ unsigned short sA[128 * 32];
  __shared__ unsigned short sW[128 * 32];
  const int m0 = blockIdx.y * 128, c0 = blockIdx.x * 128;
  const int tid = threadIdx.x, lane = tid & 31, wid = tid >> 5;
  const int waveM = wid & 1, waveN = wid >> 1;
  const int lr = lane & 15, lh = lane >> 4;
  const unsigned lds_w_off = (unsigned)(size_t)(AS3 void*)(void*)&sW[0];

  v8f acc[4][2];
#pragma unroll
  for (int i = 0; i < 4; ++i)
#pragma unroll
    for (int j = 0; j < 2; ++j) acc[i][j] = vzero8();

  for (int k0 = 0; k0 < K; k0 += 32) {
    __syncthreads();
    // W tile (128 rows x 32 halves, row stride K) via Tensor Data Mover
    if (wid == 0) {
      unsigned long long ga =
          (unsigned long long)(size_t)W + ((size_t)c0 * K + (size_t)k0) * 2ull;
      tdm_load_tile_2d(lds_w_off, ga, 32u, 128u, 32u, 128u,
                       (unsigned long long)K);
    }
    // A tile via async global->LDS copies (ASYNCcnt)
#pragma unroll
    for (int i = 0; i < 2; ++i) {
      int u = tid + i * 256;
      int row = u >> 2, kc = (u & 3) * 8;
      async_copy16(&A[(size_t)(m0 + row) * K + k0 + kc], &sA[row * 32 + kc]);
    }
    if (k0 + 32 < K)
      __builtin_prefetch(&A[(size_t)(m0 + (tid >> 1)) * K + k0 + 32], 0, 1);
    wait_async();
    if (wid == 0) wait_tensor();
    __syncthreads();

    BF16Frag bf[2];
#pragma unroll
    for (int nf = 0; nf < 2; ++nf) {
      int col = waveN * 32 + nf * 16 + lr;
      int off = col * 32 + lh * 16;  // 16 contiguous K-halves per lane
      bf[nf].u[0] = *(const u32x4*)&sW[off];
      bf[nf].u[1] = *(const u32x4*)&sW[off + 8];
    }
#pragma unroll
    for (int mf = 0; mf < 4; ++mf) {
      BF16Frag af;
      int row = waveM * 64 + mf * 16 + lr;
      int off = row * 32 + lh * 16;  // K = lh*16 + {0..15}
      af.u[0] = *(const u32x4*)&sA[off];
      af.u[1] = *(const u32x4*)&sA[off + 8];
      acc[mf][0] = wmma_bf16(af, bf[0], acc[mf][0]);
      acc[mf][1] = wmma_bf16(af, bf[1], acc[mf][1]);
    }
  }

  // Epilogue (C/D layout: M = r + 8*lh, N = lr per 16x16 tile)
#pragma unroll
  for (int mf = 0; mf < 4; ++mf)
#pragma unroll
    for (int nf = 0; nf < 2; ++nf) {
      int col = c0 + waveN * 32 + nf * 16 + lr;
      float bcol = bias[col];
#pragma unroll
      for (int r = 0; r < 8; ++r) {
        int row = m0 + waveM * 64 + mf * 16 + r + lh * 8;
        float val = acc[mf][nf][r] + bcol;
        if (mask) val *= mask[row];
        if (gate) val *= gate[(size_t)(row / batchRows) * OUT_ + (col & (OUT_ - 1))];
        if (outH) outH[(size_t)row * Ncol + col] = f2bf(val);
        else      outF[(size_t)row * Ncol + col] = val;
      }
    }
}

// ---------------------------------------------------------------------------
// Flash-style gated multi-head self-attention.
// grid (N/128, B*H, 2), 256 threads. Online softmax in f32; S and PV via WMMA.
// ---------------------------------------------------------------------------
__global__ __launch_bounds__(256) void attn_kernel(
    const unsigned short* __restrict__ vtrans, const unsigned short* __restrict__ qtrans,
    const float* __restrict__ vmask, const float* __restrict__ qmask,
    float* __restrict__ vupd, float* __restrict__ qupd) {
  const int mod = blockIdx.z;
  const unsigned short* T = mod ? qtrans : vtrans;
  const float* mk = mod ? qmask : vmask;
  float* Og = mod ? qupd : vupd;
  const int bh = blockIdx.y, b = bh >> 3, h = bh & 7;
  const int q0 = blockIdx.x * 128;
  const int kcol = h * DH_, qcol = OUT_ + h * DH_, vcol = 2 * OUT_ + h * DH_;
  const unsigned short* Tb = T + (size_t)b * N_ * TR_;
  const float* mkb = mk + (size_t)b * N_;

  __shared__ unsigned short sQ[128 * 64];   // [i][d]
  __shared__ unsigned short sK[128 * 64];   // [j][d]
  __shared__ unsigned short sVt[64 * 128];  // [d][j]
  __shared__ float          sS[128 * 128];
  __shared__ unsigned short sP[128 * 128];
  __shared__ float sM[128], sL[128], sAlpha[128];

  const int tid = threadIdx.x, lane = tid & 31, wid = tid >> 5;
  const int lr = lane & 15, lh = lane >> 4;
  const int sM0 = (wid & 1) * 64, sN0 = (wid >> 1) * 32;  // S: 2x4 waves
  const int oM0 = (wid >> 1) * 32, oN0 = (wid & 1) * 32;  // O: 4x2 waves

  // Q tile via async copies
#pragma unroll
  for (int i = 0; i < 4; ++i) {
    int u = tid + i * 256;
    int row = u >> 3, dc = (u & 7) * 8;
    async_copy16(&Tb[(size_t)(q0 + row) * TR_ + qcol + dc], &sQ[row * 64 + dc]);
  }
  if (tid < 128) { sM[tid] = -3.0e38f; sL[tid] = 0.f; }

  v8f of[2][2];
#pragma unroll
  for (int i = 0; i < 2; ++i)
#pragma unroll
    for (int j = 0; j < 2; ++j) of[i][j] = vzero8();
  wait_async();
  __syncthreads();

  for (int j0 = 0; j0 < N_; j0 += 128) {
    // K tile async; V tile transposed scatter (sync)
#pragma unroll
    for (int i = 0; i < 4; ++i) {
      int u = tid + i * 256;
      int row = u >> 3, dc = (u & 7) * 8;
      async_copy16(&Tb[(size_t)(j0 + row) * TR_ + kcol + dc], &sK[row * 64 + dc]);
      u32x4 vv = *(const u32x4*)&Tb[(size_t)(j0 + row) * TR_ + vcol + dc];
      const unsigned short* hp = (const unsigned short*)&vv;
#pragma unroll
      for (int e = 0; e < 8; ++e) sVt[(dc + e) * 128 + row] = hp[e];
    }
    wait_async();
    __syncthreads();

    // S = Q.K^T (two WMMA k-steps over d_head=64)
    v8f sacc[4][2];
#pragma unroll
    for (int i = 0; i < 4; ++i)
#pragma unroll
      for (int j = 0; j < 2; ++j) sacc[i][j] = vzero8();
#pragma unroll
    for (int kk = 0; kk < 2; ++kk) {
      BF16Frag bk[2];
#pragma unroll
      for (int nf = 0; nf < 2; ++nf) {
        int col = sN0 + nf * 16 + lr;
        int off = col * 64 + kk * 32 + lh * 16;
        bk[nf].u[0] = *(const u32x4*)&sK[off];
        bk[nf].u[1] = *(const u32x4*)&sK[off + 8];
      }
#pragma unroll
      for (int mf = 0; mf < 4; ++mf) {
        BF16Frag aq;
        int row = sM0 + mf * 16 + lr;
        int off = row * 64 + kk * 32 + lh * 16;
        aq.u[0] = *(const u32x4*)&sQ[off];
        aq.u[1] = *(const u32x4*)&sQ[off + 8];
        sacc[mf][0] = wmma_bf16(aq, bk[0], sacc[mf][0]);
        sacc[mf][1] = wmma_bf16(aq, bk[1], sacc[mf][1]);
      }
    }
    const float scale = 0.125f;  // 1/sqrt(64)
#pragma unroll
    for (int mf = 0; mf < 4; ++mf)
#pragma unroll
      for (int nf = 0; nf < 2; ++nf) {
        int col = sN0 + nf * 16 + lr;
        float mterm = (mkb[j0 + col] == 0.f) ? -3.0e38f : 0.f;
#pragma unroll
        for (int r = 0; r < 8; ++r) {
          int row = sM0 + mf * 16 + r + lh * 8;
          sS[row * 128 + col] = sacc[mf][nf][r] * scale + mterm;
        }
      }
    __syncthreads();

    // Online softmax (one thread per query row)
    if (tid < 128) {
      const int row = tid;
      float mold = sM[row], l = sL[row];
      float mmax = mold;
      for (int j = 0; j < 128; ++j) mmax = fmaxf(mmax, sS[row * 128 + j]);
      float alpha = (mold <= -1.0e38f) ? 0.f : __expf(mold - mmax);
      float lsum = 0.f;
      for (int j = 0; j < 128; ++j) {
        float p = __expf(sS[row * 128 + j] - mmax);
        lsum += p;
        sP[row * 128 + j] = f2bf(p);
      }
      sM[row] = mmax;
      sL[row] = alpha * l + lsum;
      sAlpha[row] = alpha;
    }
    __syncthreads();

    // O = alpha*O + P.V (four WMMA k-steps over 128 keys)
#pragma unroll
    for (int mf = 0; mf < 2; ++mf)
#pragma unroll
      for (int nf = 0; nf < 2; ++nf)
#pragma unroll
        for (int r = 0; r < 8; ++r) {
          int row = oM0 + mf * 16 + r + lh * 8;
          of[mf][nf][r] *= sAlpha[row];
        }
#pragma unroll
    for (int ks = 0; ks < 4; ++ks) {
      BF16Frag bv[2];
#pragma unroll
      for (int nf = 0; nf < 2; ++nf) {
        int col = oN0 + nf * 16 + lr;
        int off = col * 128 + ks * 32 + lh * 16;
        bv[nf].u[0] = *(const u32x4*)&sVt[off];
        bv[nf].u[1] = *(const u32x4*)&sVt[off + 8];
      }
#pragma unroll
      for (int mf = 0; mf < 2; ++mf) {
        BF16Frag ap;
        int row = oM0 + mf * 16 + lr;
        int off = row * 128 + ks * 32 + lh * 16;
        ap.u[0] = *(const u32x4*)&sP[off];
        ap.u[1] = *(const u32x4*)&sP[off + 8];
        of[mf][0] = wmma_bf16(ap, bv[0], of[mf][0]);
        of[mf][1] = wmma_bf16(ap, bv[1], of[mf][1]);
      }
    }
    __syncthreads();
  }

  // Normalize and write update tile
#pragma unroll
  for (int mf = 0; mf < 2; ++mf)
#pragma unroll
    for (int nf = 0; nf < 2; ++nf)
#pragma unroll
      for (int r = 0; r < 8; ++r) {
        int row = oM0 + mf * 16 + r + lh * 8;
        int col = oN0 + nf * 16 + lr;
        float l = sL[row];
        float val = (l > 0.f) ? of[mf][nf][r] / l : 0.f;
        Og[((size_t)b * N_ + q0 + row) * OUT_ + h * DH_ + col] = val;
      }
}

// ---------------------------------------------------------------------------
// Host-side orchestration
// ---------------------------------------------------------------------------
extern "C" void kernel_launch(void* const* d_in, const int* in_sizes, int n_in,
                              void* d_out, int out_size, void* d_ws, size_t ws_size,
                              hipStream_t stream) {
  (void)in_sizes; (void)n_in; (void)out_size; (void)ws_size;
  const float* v      = (const float*)d_in[0];
  const float* q      = (const float*)d_in[1];
  const float* v_mask = (const float*)d_in[2];
  const float* q_mask = (const float*)d_in[3];
  const float* w_v4q  = (const float*)d_in[4];
  const float* b_v4q  = (const float*)d_in[5];
  const float* w_q4v  = (const float*)d_in[6];
  const float* b_q4v  = (const float*)d_in[7];
  const float* w_v    = (const float*)d_in[8];
  const float* b_v    = (const float*)d_in[9];
  const float* w_q    = (const float*)d_in[10];
  const float* b_q    = (const float*)d_in[11];
  const float* w_vo   = (const float*)d_in[12];
  const float* b_vo   = (const float*)d_in[13];
  const float* w_qo   = (const float*)d_in[14];
  const float* b_qo   = (const float*)d_in[15];
  float* out = (float*)d_out;

  char* base = (char*)d_ws;
  size_t off = 0;
  auto alloc = [&](size_t bytes) -> void* {
    void* p = base + off;
    off += (bytes + 255) & ~(size_t)255;
    return p;
  };
  unsigned short* v_h    = (unsigned short*)alloc((size_t)MN_ * D_ * 2);
  unsigned short* q_h    = (unsigned short*)alloc((size_t)MN_ * D_ * 2);
  unsigned short* w_v_h  = (unsigned short*)alloc((size_t)TR_ * D_ * 2);
  unsigned short* w_q_h  = (unsigned short*)alloc((size_t)TR_ * D_ * 2);
  unsigned short* w_vo_h = (unsigned short*)alloc((size_t)OUT_ * D_ * 2);
  unsigned short* w_qo_h = (unsigned short*)alloc((size_t)OUT_ * D_ * 2);
  unsigned short* v_tr   = (unsigned short*)alloc((size_t)MN_ * TR_ * 2);
  unsigned short* q_tr   = (unsigned short*)alloc((size_t)MN_ * TR_ * 2);
  float* v_upd = (float*)alloc((size_t)MN_ * OUT_ * 4);
  float* q_upd = (float*)alloc((size_t)MN_ * OUT_ * 4);
  float* means = (float*)alloc((size_t)2 * B_ * D_ * 4);
  float* gmul  = (float*)alloc((size_t)2 * B_ * OUT_ * 4);

  const int act = MN_ * D_;
  cvt_bf16_kernel<<<(act + 255) / 256, 256, 0, stream>>>(v, v_h, act);
  cvt_bf16_kernel<<<(act + 255) / 256, 256, 0, stream>>>(q, q_h, act);
  cvt_bf16_kernel<<<(TR_ * D_ + 255) / 256, 256, 0, stream>>>(w_v, w_v_h, TR_ * D_);
  cvt_bf16_kernel<<<(TR_ * D_ + 255) / 256, 256, 0, stream>>>(w_q, w_q_h, TR_ * D_);
  cvt_bf16_kernel<<<(OUT_ * D_ + 255) / 256, 256, 0, stream>>>(w_vo, w_vo_h, OUT_ * D_);
  cvt_bf16_kernel<<<(OUT_ * D_ + 255) / 256, 256, 0, stream>>>(w_qo, w_qo_h, OUT_ * D_);

  means_kernel<<<dim3(B_, 2), 256, 0, stream>>>(v, q, v_mask, q_mask, means);
  gates_kernel<<<dim3(B_, 2), 512, 0, stream>>>(means, w_v4q, b_v4q, w_q4v, b_q4v, gmul);

  gemm_wmma_kernel<<<dim3(TR_ / 128, MN_ / 128), 256, 0, stream>>>(
      v_h, w_v_h, b_v, gmul,             v_mask, nullptr, v_tr, MN_, TR_, D_, N_);
  gemm_wmma_kernel<<<dim3(TR_ / 128, MN_ / 128), 256, 0, stream>>>(
      q_h, w_q_h, b_q, gmul + B_ * OUT_, q_mask, nullptr, q_tr, MN_, TR_, D_, N_);

  attn_kernel<<<dim3(N_ / 128, B_ * H_, 2), 256, 0, stream>>>(
      v_tr, q_tr, v_mask, q_mask, v_upd, q_upd);

  unsigned short* a_v = v_tr;  // reuse trans buffers
  unsigned short* a_q = q_tr;
  add_cvt_bf16_kernel<<<(act + 255) / 256, 256, 0, stream>>>(v, v_upd, a_v, act);
  add_cvt_bf16_kernel<<<(act + 255) / 256, 256, 0, stream>>>(q, q_upd, a_q, act);

  gemm_wmma_kernel<<<dim3(OUT_ / 128, MN_ / 128), 256, 0, stream>>>(
      a_v, w_vo_h, b_vo, nullptr, nullptr, out,                      nullptr,
      MN_, OUT_, D_, N_);
  gemm_wmma_kernel<<<dim3(OUT_ / 128, MN_ / 128), 256, 0, stream>>>(
      a_q, w_qo_h, b_qo, nullptr, nullptr, out + (size_t)MN_ * OUT_, nullptr,
      MN_, OUT_, D_, N_);
}
